// Attention_23476291239983
// MI455X (gfx1250) — matrix operand order repaired and checked by reading, not compiled, not run on hardware
//
#include <hip/hip_runtime.h>
#include <math.h>

// ---------------------------------------------------------------------------
// CDNA5 (gfx1250) wave32 WMMA types
// ---------------------------------------------------------------------------
typedef __bf16 bf16;
typedef bf16  v16bf  __attribute__((ext_vector_type(16)));
typedef bf16  bf16x8 __attribute__((ext_vector_type(8)));
typedef float v8f    __attribute__((ext_vector_type(8)));

#define KEY_SCALE 0.17677669529663687f   // 32^-0.5

__device__ __forceinline__ v8f wmma_bf16(v16bf a, v16bf b, v8f c) {
  // D = A(16x32 bf16) * B(32x16 bf16) + C(16x16 f32)
  return __builtin_amdgcn_wmma_f32_16x16x32_bf16(
      /*neg_a=*/false, a, /*neg_b=*/false, b,
      /*c_mod=*/(short)0, c, /*reuse_a=*/false, /*reuse_b=*/false);
}

// A-fragment from two contiguous bf16 runs of 8 (per-lane K = {base..base+15})
__device__ __forceinline__ v16bf a_frag_bf16(const bf16* lo) {
  bf16x8 l = *(const bf16x8*)lo;
  bf16x8 h = *(const bf16x8*)(lo + 8);
  v16bf r;
#pragma unroll
  for (int j = 0; j < 8; ++j) { r[j] = l[j]; r[8 + j] = h[j]; }
  return r;
}

// A-fragment from f32 memory (convert to bf16)
__device__ __forceinline__ v16bf a_frag_f32(const float* ap) {
  float4 a0 = *(const float4*)(ap);
  float4 a1 = *(const float4*)(ap + 4);
  float4 a2 = *(const float4*)(ap + 8);
  float4 a3 = *(const float4*)(ap + 12);
  v16bf r;
  r[0]=(bf16)a0.x;  r[1]=(bf16)a0.y;  r[2]=(bf16)a0.z;  r[3]=(bf16)a0.w;
  r[4]=(bf16)a1.x;  r[5]=(bf16)a1.y;  r[6]=(bf16)a1.z;  r[7]=(bf16)a1.w;
  r[8]=(bf16)a2.x;  r[9]=(bf16)a2.y;  r[10]=(bf16)a2.z; r[11]=(bf16)a2.w;
  r[12]=(bf16)a3.x; r[13]=(bf16)a3.y; r[14]=(bf16)a3.z; r[15]=(bf16)a3.w;
  return r;
}

// ---------------------------------------------------------------------------
// Kernel 1: weight conversion to bf16 (KEY_SCALE folded into wq)
// dst layout: [wq*s, wk, wv, wg, wo] each 128x128 row-major [out,in]
// ---------------------------------------------------------------------------
__global__ void wcvt_kernel(const float* __restrict__ wq, const float* __restrict__ wk,
                            const float* __restrict__ wv, const float* __restrict__ wg,
                            const float* __restrict__ wo, bf16* __restrict__ dst) {
  int i = blockIdx.x * 256 + threadIdx.x;
  int mat = i >> 14, idx = i & 16383;
  const float* s; float sc = 1.0f;
  switch (mat) {
    case 0: s = wq; sc = KEY_SCALE; break;
    case 1: s = wk; break;
    case 2: s = wv; break;
    case 3: s = wg; break;
    default: s = wo; break;
  }
  dst[i] = (bf16)(s[idx] * sc);
}

// ---------------------------------------------------------------------------
// Kernel 2: fused projections.
//   blockIdx.y==0 : read q_data tile once -> q (scaled, via wq*s) and g=sigmoid(.+bg)
//   blockIdx.y==1 : read m_data tile once -> k and v (v stored transposed)
// 256 threads = 8 waves; wave w owns 16-col tile w. Each block: 16 rows x 128 cols x2 mats.
// ---------------------------------------------------------------------------
__global__ __launch_bounds__(256) void proj_kernel(
    const float* __restrict__ qd, const float* __restrict__ md,
    const bf16* __restrict__ wAll, const float* __restrict__ bg,
    bf16* __restrict__ qb, bf16* __restrict__ kb,
    bf16* __restrict__ vT, bf16* __restrict__ gb) {
  const int lane = threadIdx.x & 31;
  const int ct   = threadIdx.x >> 5;   // col tile 0..7
  const int rt   = blockIdx.x;         // row tile 0..4095
  const int z    = blockIdx.y;         // 0: q&g from q_data, 1: k&v from m_data
  const int l15  = lane & 15;
  const int hiA  = (lane >= 16) ? 16 : 0;
  const int hiB  = (lane >= 16) ? 16 : 0;
  const int rOff = (lane >= 16) ? 8 : 0;

  const float* src = (z == 0) ? qd : md;
  const bf16* W1 = wAll + (size_t)((z == 0) ? 0 : 1) * 16384;  // wq*s or wk
  const bf16* W2 = wAll + (size_t)((z == 0) ? 3 : 2) * 16384;  // wg   or wv
  const int row = rt * 16 + l15;
  const int col = ct * 16 + l15;

  v8f c1, c2;
#pragma unroll
  for (int r = 0; r < 8; ++r) { c1[r] = 0.f; c2[r] = 0.f; }

#pragma unroll
  for (int ks = 0; ks < 4; ++ks) {
    v16bf a = a_frag_f32(src + (size_t)row * 128 + ks * 32 + hiA);
    v16bf b1 = *(const v16bf*)(W1 + (size_t)col * 128 + ks * 32 + hiB);
    v16bf b2 = *(const v16bf*)(W2 + (size_t)col * 128 + ks * 32 + hiB);
    c1 = wmma_bf16(a, b1, c1);
    c2 = wmma_bf16(a, b2, c2);
  }

#pragma unroll
  for (int r = 0; r < 8; ++r) {
    const int m = r + rOff;
    const size_t token = (size_t)rt * 16 + m;
    if (z == 0) {
      qb[token * 128 + col] = (bf16)c1[r];
      float gv = 1.0f / (1.0f + __expf(-(c2[r] + bg[col])));
      gb[token * 128 + col] = (bf16)gv;
    } else {
      kb[token * 128 + col] = (bf16)c1[r];
      const int n1 = (int)(token >> 8), kpos = (int)(token & 255);
      const int h = col >> 5, dh = col & 31;
      vT[((size_t)(n1 * 4 + h) * 32 + dh) * 256 + kpos] = (bf16)c2[r];
    }
  }
}

// ---------------------------------------------------------------------------
// Kernel 3: flash attention per (n1, h). 8 waves x 2 q-tiles = 256 q rows.
// DH=32 -> QK^T is a single 16x16x32 WMMA per (q-tile, k-tile).
// Online softmax; P re-laid-out C/D->A via per-wave LDS staging (in-order DS).
// Epilogue applies 1/l and the sigmoid gate, writes wa in [token, C] bf16.
// ---------------------------------------------------------------------------
__global__ __launch_bounds__(256) void attn_kernel(
    const bf16* __restrict__ qb, const bf16* __restrict__ kb,
    const bf16* __restrict__ vT, const bf16* __restrict__ gb,
    const float* __restrict__ bias, const float* __restrict__ nbb,
    bf16* __restrict__ wab) {
  __shared__ __align__(32) bf16 pstage[8][16 * 32];
  const int lane = threadIdx.x & 31;
  const int wid  = threadIdx.x >> 5;
  const int n1 = blockIdx.x, h = blockIdx.y;
  const int l15  = lane & 15;
  const int hiA  = (lane >= 16) ? 16 : 0;
  const int hiB  = (lane >= 16) ? 16 : 0;
  const int rOff = (lane >= 16) ? 8 : 0;
  bf16* ps = pstage[wid];
  const bf16* vhead = vT + (size_t)(n1 * 4 + h) * 32 * 256;

  for (int qi = 0; qi < 2; ++qi) {
    const int qt = wid * 2 + qi;
    const int qpos = qt * 16 + l15;
    v16bf aq = a_frag_bf16(qb + ((size_t)(n1 * 256 + qpos)) * 128 + h * 32 + hiA);

    v8f O0, O1;
    float mrow[8], lrow[8];
#pragma unroll
    for (int r = 0; r < 8; ++r) { O0[r] = 0.f; O1[r] = 0.f; mrow[r] = -3.0e38f; lrow[r] = 0.f; }

    for (int kt2 = 0; kt2 < 8; ++kt2) {
      v8f s[2];
#pragma unroll
      for (int t = 0; t < 2; ++t) {
        const int kt = kt2 * 2 + t;
        const int kpos = kt * 16 + l15;
        v16bf bk = *(const v16bf*)(kb + ((size_t)(n1 * 256 + kpos)) * 128 + h * 32 + hiB);
        v8f zc;
#pragma unroll
        for (int r = 0; r < 8; ++r) zc[r] = 0.f;
        v8f sv = wmma_bf16(aq, bk, zc);
        const float bv = bias[n1 * 256 + kpos];
#pragma unroll
        for (int r = 0; r < 8; ++r) {
          const int qr = qt * 16 + r + rOff;
          sv[r] += bv + nbb[((size_t)(h * 256 + qr)) * 256 + kpos];
        }
        s[t] = sv;
      }

      // online softmax stats (rows live across 16-lane halves; xor 1,2,4,8 stays in-half)
#pragma unroll
      for (int r = 0; r < 8; ++r) {
        float mx = fmaxf(s[0][r], s[1][r]);
#pragma unroll
        for (int o = 1; o < 16; o <<= 1) mx = fmaxf(mx, __shfl_xor(mx, o, 32));
        const float mn = fmaxf(mrow[r], mx);
        const float sc = __expf(mrow[r] - mn);
        const float p0 = __expf(s[0][r] - mn);
        const float p1 = __expf(s[1][r] - mn);
        float rs = p0 + p1;
#pragma unroll
        for (int o = 1; o < 16; o <<= 1) rs += __shfl_xor(rs, o, 32);
        lrow[r] = lrow[r] * sc + rs;
        mrow[r] = mn;
        O0[r] *= sc; O1[r] *= sc;
        const int m = r + rOff;
        ps[m * 32 + l15]      = (bf16)p0;
        ps[m * 32 + 16 + l15] = (bf16)p1;
      }

      // C/D layout -> A layout through LDS (same-wave DS ops are in-order)
      v16bf pa = a_frag_bf16(ps + l15 * 32 + hiA);
      const bf16* vb = vhead + kt2 * 32 + hiB;
      v16bf bv0 = *(const v16bf*)(vb + (size_t)l15 * 256);
      v16bf bv1 = *(const v16bf*)(vb + (size_t)(16 + l15) * 256);
      O0 = wmma_bf16(pa, bv0, O0);
      O1 = wmma_bf16(pa, bv1, O1);
    }

#pragma unroll
    for (int r = 0; r < 8; ++r) {
      const float inv = 1.0f / lrow[r];
      const size_t token = (size_t)n1 * 256 + qt * 16 + r + rOff;
      const int c0 = h * 32 + l15;
      const float g0 = (float)gb[token * 128 + c0];
      const float g1 = (float)gb[token * 128 + c0 + 16];
      wab[token * 128 + c0]      = (bf16)(O0[r] * inv * g0);
      wab[token * 128 + c0 + 16] = (bf16)(O1[r] * inv * g1);
    }
  }
}

// ---------------------------------------------------------------------------
// Kernel 4: output projection  out = wa @ wo^T + bo   (f32 output)
// ---------------------------------------------------------------------------
__global__ __launch_bounds__(256) void oproj_kernel(
    const bf16* __restrict__ wab, const bf16* __restrict__ woBf,
    const float* __restrict__ bo, float* __restrict__ out) {
  const int lane = threadIdx.x & 31;
  const int ct   = threadIdx.x >> 5;
  const int rt   = blockIdx.x;
  const int l15  = lane & 15;
  const int hiA  = (lane >= 16) ? 16 : 0;
  const int hiB  = (lane >= 16) ? 16 : 0;
  const int rOff = (lane >= 16) ? 8 : 0;
  const int row = rt * 16 + l15;
  const int col = ct * 16 + l15;

  v8f c;
#pragma unroll
  for (int r = 0; r < 8; ++r) c[r] = 0.f;

#pragma unroll
  for (int ks = 0; ks < 4; ++ks) {
    v16bf a = a_frag_bf16(wab + (size_t)row * 128 + ks * 32 + hiA);
    v16bf b = *(const v16bf*)(woBf + (size_t)col * 128 + ks * 32 + hiB);
    c = wmma_bf16(a, b, c);
  }

  const float bv = bo[col];
#pragma unroll
  for (int r = 0; r < 8; ++r) {
    const size_t token = (size_t)rt * 16 + r + rOff;
    out[token * 128 + col] = c[r] + bv;
  }
}

// ---------------------------------------------------------------------------
// Host launcher
// ---------------------------------------------------------------------------
extern "C" void kernel_launch(void* const* d_in, const int* in_sizes, int n_in,
                              void* d_out, int out_size, void* d_ws, size_t ws_size,
                              hipStream_t stream) {
  (void)in_sizes; (void)n_in; (void)out_size; (void)ws_size;
  const float* qd   = (const float*)d_in[0];
  const float* md   = (const float*)d_in[1];
  const float* bias = (const float*)d_in[2];
  const float* nbb  = (const float*)d_in[3];
  const float* wq   = (const float*)d_in[4];
  const float* wk   = (const float*)d_in[5];
  const float* wv   = (const float*)d_in[6];
  const float* wo   = (const float*)d_in[7];
  const float* bo   = (const float*)d_in[8];
  const float* wg   = (const float*)d_in[9];
  const float* bg   = (const float*)d_in[10];
  float* out = (float*)d_out;

  bf16* ws = (bf16*)d_ws;
  const size_t WSZ = (size_t)5 * 128 * 128;     // 5 weight matrices
  const size_t TC  = (size_t)65536 * 128;       // one [T, C] bf16 buffer
  bf16* wAll = ws;
  bf16* qb   = ws + WSZ;
  bf16* kbuf = qb + TC;
  bf16* vT   = kbuf + TC;
  bf16* gb   = vT + TC;
  bf16* wab  = gb + TC;

  wcvt_kernel<<<320, 256, 0, stream>>>(wq, wk, wv, wg, wo, wAll);
  proj_kernel<<<dim3(4096, 2), 256, 0, stream>>>(qd, md, wAll, bg, qb, kbuf, vT, gb);
  attn_kernel<<<dim3(256, 4), 256, 0, stream>>>(qb, kbuf, vT, gb, bias, nbb, wab);
  oproj_kernel<<<4096, 256, 0, stream>>>(wab, wAll + (size_t)4 * 16384, bo, out);
}
